// GNN_WithMetadata_86294482911416
// MI455X (gfx1250) — matrix operand order repaired and checked by reading, not compiled, not run on hardware
//
#include <hip/hip_runtime.h>
#include <hip/hip_bf16.h>
#include <math.h>

typedef __attribute__((ext_vector_type(16))) _Float16 v16h;
typedef __attribute__((ext_vector_type(8)))  _Float16 v8h;
typedef __attribute__((ext_vector_type(8)))  float    v8f;

#define NN    100000
#define NE    640000
#define EF    (NE + NN)
#define HID   128
#define HEADS 4
#define BGR   64
#define METsocial 16
#define META  16
#define MID   64
#define NEG_SLOPE 0.2f
#define BN_EPS    1e-5f

// ---------------- utility ----------------
__device__ inline void atomicMaxF(float* addr, float val) {
    unsigned int* ua = (unsigned int*)addr;
    unsigned int old = *ua;
    while (__uint_as_float(old) < val) {
        unsigned int assumed = old;
        old = atomicCAS(ua, assumed, __float_as_uint(val));
        if (old == assumed) break;
    }
}

__global__ void k_fill(float* __restrict__ p, float v, int n) {
    int i = blockIdx.x * blockDim.x + threadIdx.x;
    if (i < n) p[i] = v;
}

// h[n,c] = x[n]*node_w[c] + node_b[c]   (rank-1 node encoder)
__global__ void k_init_h(const float* __restrict__ x, const float* __restrict__ nw,
                         const float* __restrict__ nb, float* __restrict__ h) {
    int i = blockIdx.x * blockDim.x + threadIdx.x;
    if (i >= NN * HID) return;
    int n = i >> 7, c = i & 127;
    h[i] = x[n] * nw[c] + nb[c];
}

// deg[dst]+=1 ; S[dst]+=edge_attr
__global__ void k_degS(const int* __restrict__ ei, const float* __restrict__ ea,
                       float* __restrict__ degf, float* __restrict__ S) {
    int e = blockIdx.x * blockDim.x + threadIdx.x;
    if (e >= NE) return;
    int d = ei[NE + e];
    atomicAdd(&degf[d], 1.0f);
    atomicAdd(&S[d], ea[e]);
}

// u[n]=S/max(deg,1) ; v[n]=deg/max(deg,1)  (self-loop attr is rank-1 in these)
__global__ void k_uv(const float* __restrict__ degf, const float* __restrict__ S,
                     float* __restrict__ u, float* __restrict__ v) {
    int n = blockIdx.x * blockDim.x + threadIdx.x;
    if (n >= NN) return;
    float dg = degf[n], m = fmaxf(dg, 1.0f);
    u[n] = S[n] / m;
    v[n] = dg / m;
}

// fold a_e GEMM to scalars: c1[l][h]=edge_w . (lin_edge_w att_edge), c0 with edge_b
// c01 layout: [l*8 + h] = c0 ; [l*8 + 4 + h] = c1
__global__ void k_consts(const float* __restrict__ lew, const float* __restrict__ ae,
                         const float* __restrict__ ew, const float* __restrict__ eb,
                         float* __restrict__ c01) {
    int t = threadIdx.x;
    if (t >= 8) return;
    int l = t >> 2, hh = t & 3;
    float c0 = 0.f, c1 = 0.f;
    for (int k = 0; k < 128; ++k) {
        float we = 0.f;
        for (int d = 0; d < 32; ++d)
            we += lew[l * 16384 + k * 128 + hh * 32 + d] * ae[l * 128 + hh * 32 + d];
        c1 += ew[k] * we;
        c0 += eb[k] * we;
    }
    c01[l * 8 + hh]     = c0;
    c01[l * 8 + 4 + hh] = c1;
}

__global__ void k_convw(const float* __restrict__ w, _Float16* __restrict__ wh, int n) {
    int i = blockIdx.x * blockDim.x + threadIdx.x;
    if (i < n) wh[i] = (_Float16)w[i];
}

// ---------------- WMMA GEMM: xh = h @ W (128x128), f16 inputs, f32 accum ---------
// block = 128 threads (4 waves); 16 output rows per block; wave w -> cols [32w,32w+32)
__global__ void k_gemm(const float* __restrict__ h, const _Float16* __restrict__ Wh,
                       float* __restrict__ xh) {
    __shared__ _Float16 As[16 * 136];         // padded: stride 136 halves kills bank conflicts
    const int row0 = blockIdx.x * 16;
    const int t = threadIdx.x;
    for (int i = 0; i < 16; ++i) {            // stage 16x128 f32 rows -> f16 LDS
        int idx = t + i * 128;
        int r = idx >> 7, c = idx & 127;
        As[r * 136 + c] = (_Float16)h[(row0 + r) * 128 + c];
    }
    __syncthreads();

    const int lane = t & 31, wave = t >> 5;
    const int nbase = wave * 32;
    v8f acc0 = {}; v8f acc1 = {};
    const int m = lane & 15;
    const int koff = (lane < 16) ? 0 : 8;     // ISA 16-bit A layout: lanes 16-31 carry K+8
    for (int kc = 0; kc < 4; ++kc) {
        const int kb = kc * 32;
        v8h lo = *(const v8h*)&As[m * 136 + kb + koff];        // K = kb+koff .. +7
        v8h hi = *(const v8h*)&As[m * 136 + kb + 16 + koff];   // K = kb+16+koff .. +7
        v16h a = __builtin_shufflevector(lo, hi, 0,1,2,3,4,5,6,7,8,9,10,11,12,13,14,15);
        v16h b0, b1;                                                   // lane->N, elem->K (same K order as A)
        for (int s = 0; s < 16; ++s) {
            const int kk = kb + koff + (s < 8 ? s : 8 + s);
            b0[s] = Wh[kk * 128 + nbase + m];
            b1[s] = Wh[kk * 128 + nbase + 16 + m];
        }
        acc0 = __builtin_amdgcn_wmma_f32_16x16x32_f16(false, a, false, b0, (short)0, acc0, false, false);
        acc1 = __builtin_amdgcn_wmma_f32_16x16x32_f16(false, a, false, b1, (short)0, acc1, false, false);
    }
    const int nc = lane & 15, mb = (lane >> 4) * 8;   // C/D layout: VGPR r -> M=r(+8), lane -> N
    for (int r = 0; r < 8; ++r) {
        xh[(row0 + mb + r) * 128 + nbase + nc]      = acc0[r];
        xh[(row0 + mb + r) * 128 + nbase + 16 + nc] = acc1[r];
    }
}

// a_s[n,h], a_d[n,h]: per-head 32-dot of xh row; one node per block, wave w == head w
__global__ void k_attn(const float* __restrict__ xh, const float* __restrict__ as_w,
                       const float* __restrict__ ad_w, float* __restrict__ a_s,
                       float* __restrict__ a_d) {
    int n = blockIdx.x;
    int c = threadIdx.x;
    int head = c >> 5, lane = c & 31;
    float xv = xh[n * 128 + c];
    float s = xv * as_w[head * 32 + lane];
    float d = xv * ad_w[head * 32 + lane];
    for (int off = 16; off > 0; off >>= 1) {
        s += __shfl_down(s, off, 32);
        d += __shfl_down(d, off, 32);
    }
    if (lane == 0) { a_s[n * 4 + head] = s; a_d[n * 4 + head] = d; }
}

// alpha = leaky_relu(a_s[src]+a_d[dst]+a_e) and segment max via CAS float-max
__global__ void k_alpha(const int* __restrict__ ei, const float* __restrict__ ea,
                        const float* __restrict__ u, const float* __restrict__ v,
                        const float* __restrict__ a_s, const float* __restrict__ a_d,
                        const float* __restrict__ cl, float* __restrict__ alpha,
                        float* __restrict__ amax) {
    int f = blockIdx.x * blockDim.x + threadIdx.x;
    if (f >= EF) return;
    int s, d; float wa, wb;
    if (f < NE) { s = ei[f]; d = ei[NE + f]; wa = ea[f]; wb = 1.0f; }
    else        { int n = f - NE; s = n; d = n; wa = u[n]; wb = v[n]; }
    #pragma unroll
    for (int h = 0; h < 4; ++h) {
        float al = a_s[s * 4 + h] + a_d[d * 4 + h] + wa * cl[4 + h] + wb * cl[h];
        al = (al > 0.f) ? al : NEG_SLOPE * al;
        alpha[f * 4 + h] = al;
        atomicMaxF(&amax[d * 4 + h], al);
    }
}

__global__ void k_expden(const int* __restrict__ ei, float* __restrict__ alpha,
                         const float* __restrict__ amax, float* __restrict__ den) {
    int f = blockIdx.x * blockDim.x + threadIdx.x;
    if (f >= EF) return;
    int d = (f < NE) ? ei[NE + f] : (f - NE);
    #pragma unroll
    for (int h = 0; h < 4; ++h) {
        float m = amax[d * 4 + h];
        if (!(m > -INFINITY)) m = 0.0f;      // mirror where(isfinite, amax, 0)
        float ex = __expf(alpha[f * 4 + h] - m);
        alpha[f * 4 + h] = ex;
        atomicAdd(&den[d * 4 + h], ex);
    }
}

// out[dst,:] += xh[src,:] * w   — one wave per flat edge, float4 per lane
__global__ void k_scatter(const int* __restrict__ ei, const float* __restrict__ ex,
                          const float* __restrict__ den, const float* __restrict__ xh,
                          float* __restrict__ outa) {
    int f = blockIdx.x * 8 + (threadIdx.x >> 5);
    if (f >= EF) return;
    int lane = threadIdx.x & 31;
    int s, d;
    if (f < NE) { s = ei[f]; d = ei[NE + f]; }
    else        { s = f - NE; d = s; }
    int head = lane >> 3;                     // channels 4*lane..+3 are all in head lane>>3
    float w = ex[f * 4 + head] / (den[d * 4 + head] + 1e-16f);
    const float4 xv = *(const float4*)&xh[s * 128 + lane * 4];
    float* o = &outa[d * 128 + lane * 4];
    atomicAdd(o + 0, xv.x * w);
    atomicAdd(o + 1, xv.y * w);
    atomicAdd(o + 2, xv.z * w);
    atomicAdd(o + 3, xv.w * w);
}

// bias + BN(eval) + relu + residual, in place into h
__global__ void k_post(const float* __restrict__ outa, const float* __restrict__ bias,
                       const float* __restrict__ g, const float* __restrict__ b,
                       const float* __restrict__ mu, const float* __restrict__ var,
                       float* __restrict__ h) {
    int i = blockIdx.x * blockDim.x + threadIdx.x;
    if (i >= NN * HID) return;
    int c = i & 127;
    float o = outa[i] + bias[c];
    o = g[c] * (o - mu[c]) * rsqrtf(var[c] + BN_EPS) + b[c];
    h[i] = fmaxf(o, 0.0f) + h[i];
}

__global__ void k_pool(const float* __restrict__ h, const int* __restrict__ batch,
                       float* __restrict__ pooled, float* __restrict__ cnt) {
    int i = blockIdx.x * blockDim.x + threadIdx.x;
    if (i >= NN * HID) return;
    int n = i >> 7, c = i & 127;
    int bg = batch[n];
    atomicAdd(&pooled[bg * 128 + c], h[i]);
    if (c == 0) atomicAdd(&cnt[bg], 1.0f);
}

// tiny classifier: one thread per graph
__global__ void k_cls(const float* __restrict__ pooled, const float* __restrict__ cnt,
                      const float* __restrict__ meta, const float* __restrict__ w1,
                      const float* __restrict__ b1, const float* __restrict__ w2,
                      const float* __restrict__ b2, float* __restrict__ out) {
    int gph = threadIdx.x;
    if (gph >= BGR) return;
    float inv = 1.0f / fmaxf(cnt[gph], 1.0f);
    float logit = b2[0];
    for (int j = 0; j < MID; ++j) {
        float acc = b1[j];
        for (int i = 0; i < HID; ++i)
            acc += pooled[gph * 128 + i] * inv * w1[i * MID + j];
        for (int i = 0; i < META; ++i)
            acc += meta[gph * META + i] * w1[(HID + i) * MID + j];
        logit += fmaxf(acc, 0.0f) * w2[j];
    }
    out[gph] = 1.0f / (1.0f + __expf(-logit));
}

// ---------------- host launch ----------------
extern "C" void kernel_launch(void* const* d_in, const int* in_sizes, int n_in,
                              void* d_out, int out_size, void* d_ws, size_t ws_size,
                              hipStream_t stream) {
    const float* x     = (const float*)d_in[0];
    const float* eattr = (const float*)d_in[1];
    const float* meta  = (const float*)d_in[2];
    const int*   ei    = (const int*)  d_in[3];
    const int*   batch = (const int*)  d_in[4];
    const float* node_w = (const float*)d_in[5];
    const float* node_b = (const float*)d_in[6];
    const float* edge_w = (const float*)d_in[7];
    const float* edge_b = (const float*)d_in[8];
    const float* lin_w  = (const float*)d_in[9];
    const float* lin_edge_w = (const float*)d_in[10];
    const float* att_src  = (const float*)d_in[11];
    const float* att_dst  = (const float*)d_in[12];
    const float* att_edge = (const float*)d_in[13];
    const float* conv_bias = (const float*)d_in[14];
    const float* bn_g  = (const float*)d_in[15];
    const float* bn_b  = (const float*)d_in[16];
    const float* bn_m  = (const float*)d_in[17];
    const float* bn_v  = (const float*)d_in[18];
    const float* w1 = (const float*)d_in[19];
    const float* b1 = (const float*)d_in[20];
    const float* w2 = (const float*)d_in[21];
    const float* b2 = (const float*)d_in[22];
    float* out = (float*)d_out;

    // workspace carve-up (256B aligned)
    char* base = (char*)d_ws;
    size_t off = 0;
    auto carve = [&](size_t bytes) {
        char* p = base + off;
        off += (bytes + 255) & ~(size_t)255;
        return p;
    };
    float*    h      = (float*)carve((size_t)NN * HID * 4);
    float*    xh     = (float*)carve((size_t)NN * HID * 4);
    float*    outa   = (float*)carve((size_t)NN * HID * 4);
    float*    alpha  = (float*)carve((size_t)EF * 4 * 4);
    float*    amax   = (float*)carve((size_t)NN * 4 * 4);
    float*    den    = (float*)carve((size_t)NN * 4 * 4);
    float*    a_s    = (float*)carve((size_t)NN * 4 * 4);
    float*    a_d    = (float*)carve((size_t)NN * 4 * 4);
    float*    degf   = (float*)carve((size_t)NN * 4);
    float*    Ssum   = (float*)carve((size_t)NN * 4);
    float*    u      = (float*)carve((size_t)NN * 4);
    float*    v      = (float*)carve((size_t)NN * 4);
    _Float16* wh     = (_Float16*)carve((size_t)2 * 128 * 128 * 2);
    float*    c01    = (float*)carve(16 * 4);
    float*    pooled = (float*)carve((size_t)BGR * 128 * 4);
    float*    cnt    = (float*)carve((size_t)BGR * 4);

    const int T = 256;
    auto blks = [](long n, int t) { return (unsigned)((n + t - 1) / t); };

    // ---- preamble ----
    k_fill<<<blks(NN, T), T, 0, stream>>>(degf, 0.f, NN);
    k_fill<<<blks(NN, T), T, 0, stream>>>(Ssum, 0.f, NN);
    k_fill<<<blks(BGR * 128, T), T, 0, stream>>>(pooled, 0.f, BGR * 128);
    k_fill<<<1, T, 0, stream>>>(cnt, 0.f, BGR);
    k_init_h<<<blks((long)NN * HID, T), T, 0, stream>>>(x, node_w, node_b, h);
    k_degS<<<blks(NE, T), T, 0, stream>>>(ei, eattr, degf, Ssum);
    k_uv<<<blks(NN, T), T, 0, stream>>>(degf, Ssum, u, v);
    k_consts<<<1, 8, 0, stream>>>(lin_edge_w, att_edge, edge_w, edge_b, c01);
    k_convw<<<blks(2 * 128 * 128, T), T, 0, stream>>>(lin_w, wh, 2 * 128 * 128);

    // ---- GAT layers ----
    for (int l = 0; l < 2; ++l) {
        k_fill<<<blks(NN * 4, T), T, 0, stream>>>(amax, -INFINITY, NN * 4);
        k_fill<<<blks(NN * 4, T), T, 0, stream>>>(den, 0.f, NN * 4);
        k_fill<<<blks((long)NN * HID, T), T, 0, stream>>>(outa, 0.f, NN * HID);

        k_gemm<<<NN / 16, 128, 0, stream>>>(h, wh + (size_t)l * 16384, xh);
        k_attn<<<NN, 128, 0, stream>>>(xh, att_src + l * 128, att_dst + l * 128, a_s, a_d);
        k_alpha<<<blks(EF, T), T, 0, stream>>>(ei, eattr, u, v, a_s, a_d, c01 + l * 8, alpha, amax);
        k_expden<<<blks(EF, T), T, 0, stream>>>(ei, alpha, amax, den);
        k_scatter<<<blks(EF, 8), 256, 0, stream>>>(ei, alpha, den, xh, outa);
        k_post<<<blks((long)NN * HID, T), T, 0, stream>>>(outa, conv_bias + l * 128,
                                                          bn_g + l * 128, bn_b + l * 128,
                                                          bn_m + l * 128, bn_v + l * 128, h);
    }

    // ---- pool + classifier ----
    k_pool<<<blks((long)NN * HID, T), T, 0, stream>>>(h, batch, pooled, cnt);
    k_cls<<<1, BGR, 0, stream>>>(pooled, cnt, meta, w1, b1, w2, b2, out);
}
